// Graph_Attention_79474074845611
// MI455X (gfx1250) — matrix operand order repaired and checked by reading, not compiled, not run on hardware
//
#include <hip/hip_runtime.h>

typedef __attribute__((ext_vector_type(16))) _Float16 v16h;
typedef __attribute__((ext_vector_type(8)))  _Float16 v8h;
typedef __attribute__((ext_vector_type(8)))  float    v8f;

#define T_DIM 32
#define N_DIM 2048
#define F_DIM 128
#define D_DIM 128
#define INV_SCALE 0.08838834764831845f   // 1/sqrt(128)

// ---- fragment helpers -------------------------------------------------------
// A-matrix (f16, 16x32) per CDNA5 ISA: lane<16 -> K in {0..7, 16..23},
// lane>=16 -> K in {8..15, 24..31}; half h<8 -> K=base+h, h>=8 -> K=base+16+(h-8)
static __device__ __forceinline__ v16h load_a_f32(const float* row, int f0, int base) {
    v16h a;
    const float* p0 = row + f0 + base;
    const float* p1 = p0 + 16;
#pragma unroll
    for (int h = 0; h < 8; ++h) {
        a[h]     = (_Float16)p0[h];
        a[8 + h] = (_Float16)p1[h];
    }
    return a;
}
// B-matrix (f16, 32x16): lane holds column (lane&15), K = (lane>=16?8:0)+h for h<8, +16+(h-8) for h>=8
static __device__ __forceinline__ v16h load_b_f32(const float* p) {
    v16h b;
#pragma unroll
    for (int h = 0; h < 16; ++h) b[h] = (_Float16)p[h < 8 ? h : h + 8];
    return b;
}

// ---- kernel 1: zero the temp-sum accumulator S [N,D] ------------------------
__global__ __launch_bounds__(256) void k_zero(float* S) {
    S[(size_t)blockIdx.x * 256 + threadIdx.x] = 0.0f;
}

// ---- kernel 2: compress adj (+self loops) into a bitmask, 1 word per wave ---
__global__ __launch_bounds__(256) void k_mask(const int* __restrict__ adj,
                                              unsigned* __restrict__ mask) {
    unsigned wid  = blockIdx.x * 8u + (threadIdx.x >> 5);   // word id [t][i][w]
    unsigned lane = threadIdx.x & 31u;
    unsigned t    = wid >> 17;          // / (2048*64)
    unsigned rem  = wid & 131071u;
    unsigned i    = rem >> 6;
    unsigned w    = rem & 63u;
    unsigned j    = w * 32u + lane;
    int v = adj[((size_t)t * N_DIM + i) * N_DIM + j];
    bool pred = (v != 0) || (j == i);   // adj + I, clamped
    unsigned long long bal = __ballot(pred);
    if (lane == 0) mask[wid] = (unsigned)bal;
}

// ---- kernel 3: E[t,n] = exp(f2[t,n]/sqrt(D)), f2 collapsed to a mat-vec -----
__global__ __launch_bounds__(256) void k_e(const float* __restrict__ inp,
                                           const float* __restrict__ Wk,
                                           const float* __restrict__ bk,
                                           const float* __restrict__ f2w,
                                           const float* __restrict__ f2b,
                                           float* __restrict__ E) {
    __shared__ float wk2[F_DIM];
    int tid = threadIdx.x;
    if (tid < F_DIM) {                       // wk2[f] = sum_d Wk[d][f]*f2w[d]
        float s = 0.f;
        for (int d = 0; d < D_DIM; ++d) s = fmaf(Wk[d * F_DIM + tid], f2w[d], s);
        wk2[tid] = s;
    }
    __syncthreads();
    int gid = blockIdx.x * 256 + tid;        // t*N + n
    int n   = gid & (N_DIM - 1);
    const float* row = inp + (size_t)gid * F_DIM;
    float s = 0.f;
    for (int f = 0; f < F_DIM; ++f) s = fmaf(row[f], wk2[f], s);
    const float* bkr = bk + (size_t)n * D_DIM;
    for (int d = 0; d < D_DIM; ++d) s = fmaf(bkr[d], f2w[d], s);
    s += f2b[0];
    E[gid] = expf(s * INV_SCALE);
}

// ---- kernel 4: V projection via WMMA, stored transposed in f16: vT[t][d][n] -
__global__ __launch_bounds__(256) void k_vproj(const float* __restrict__ inp,
                                               const float* __restrict__ Wv,
                                               const float* __restrict__ bv,
                                               _Float16* __restrict__ vT) {
    int wave = threadIdx.x >> 5, lane = threadIdx.x & 31;
    int tile = blockIdx.x * 8 + wave;        // 0..4095 : (t, 16-row n tile)
    int t = tile >> 7, n0 = (tile & 127) * 16;
    int nlo = lane & 15, hi = lane >> 4, base = hi * 8;

    const float* arow = inp + ((size_t)t * N_DIM + n0 + nlo) * F_DIM;
    v16h A[4];
#pragma unroll
    for (int kt = 0; kt < 4; ++kt) A[kt] = load_a_f32(arow, kt * 32, base);

#pragma unroll
    for (int dt = 0; dt < 8; ++dt) {
        v8f acc = {};
        const float* brow = Wv + (size_t)(dt * 16 + nlo) * F_DIM + hi * 8;
#pragma unroll
        for (int kt = 0; kt < 4; ++kt) {
            v16h B = load_b_f32(brow + kt * 32);
            acc = __builtin_amdgcn_wmma_f32_16x16x32_f16(false, A[kt], false, B,
                                                         (short)0, acc, false, false);
        }
        const float* bvp = bv + (size_t)(n0 + 8 * hi) * D_DIM + dt * 16 + nlo;
        v8h o;
#pragma unroll
        for (int c = 0; c < 8; ++c) o[c] = (_Float16)(acc[c] + bvp[c * D_DIM]);
        *(v8h*)(vT + ((size_t)t * D_DIM + dt * 16 + nlo) * N_DIM + n0 + 8 * hi) = o;
    }
}

// ---- kernel 5: masked attention + attn@V via WMMA, t-sum into S (atomics) ---
__global__ __launch_bounds__(256) void k_attn(const float* __restrict__ E,
                                              const unsigned* __restrict__ mask,
                                              const _Float16* __restrict__ vT,
                                              float* __restrict__ S) {
    __shared__ float Elds[N_DIM];
    int t = blockIdx.x >> 4, grp = blockIdx.x & 15;
    int tid = threadIdx.x, wave = tid >> 5, lane = tid & 31;

    { // stage E[t,:] (2048 f32) via gfx1250 async global->LDS DMA (ASYNCcnt)
        unsigned lds_off = (unsigned)(size_t)(&Elds[0]) + (unsigned)tid * 16u;
        unsigned long long gaddr =
            (unsigned long long)(size_t)((const char*)(E + (size_t)t * N_DIM) + tid * 16);
        asm volatile("global_load_async_to_lds_b128 %0, %1, off"
                     :: "v"(lds_off), "v"(gaddr) : "memory");
        asm volatile("global_load_async_to_lds_b128 %0, %1, off offset:4096"
                     :: "v"(lds_off), "v"(gaddr) : "memory");
        asm volatile("s_wait_asynccnt 0" ::: "memory");
    }
    __syncthreads();

    int i0 = (grp * 8 + wave) * 16;
    int r = lane & 15, hi = lane >> 4, base = hi * 16;
    const unsigned* mrow = mask + ((size_t)t * N_DIM + i0 + r) * 64;
    __builtin_prefetch(mrow, 0, 1);                  // global_prefetch_b8
    __builtin_prefetch(mrow + 32, 0, 1);

    // pass 1: row sum s_r = sum_j mask * E_j  (lane pair splits j range)
    float s = 0.f;
    {
        int wstart = hi * 32, jb = hi * 1024;
        for (int w = 0; w < 32; ++w) {
            unsigned word = mrow[wstart + w];
            const float4* ep = (const float4*)(Elds + jb + w * 32);
            float ev[32];
#pragma unroll
            for (int q = 0; q < 8; ++q) *(float4*)(ev + q * 4) = ep[q];
#pragma unroll
            for (int b = 0; b < 32; ++b)
                s += ((word >> b) & 1u) ? ev[b] : 0.0f;
        }
    }
    s += __shfl_xor(s, 16);
    float inv_s = 1.0f / s;

    // pass 2: attn tile in registers -> WMMA against vT
    v8f acc[8] = {};
    const _Float16* vbase = vT + (size_t)t * D_DIM * N_DIM;
    for (int step = 0; step < 64; ++step) {
        int j0 = step * 32;
        unsigned word = mrow[step];
        if ((step & 7) == 0) __builtin_prefetch(mrow + step + 8, 0, 1);
        // two contiguous 8-float chunks per lane -> ds_load_b128 x4
        float c0[8], c1[8];
        {
            const float4* e0 = (const float4*)(Elds + j0 + base);
            const float4* e1 = (const float4*)(Elds + j0 + base + 8);
            *(float4*)(c0 + 0) = e0[0]; *(float4*)(c0 + 4) = e0[1];
            *(float4*)(c1 + 0) = e1[0]; *(float4*)(c1 + 4) = e1[1];
        }
        v16h A;
#pragma unroll
        for (int h = 0; h < 8; ++h) {
            A[h]     = (_Float16)(((word >> (base + h))      & 1u) ? c0[h] * inv_s : 0.0f);
            A[8 + h] = (_Float16)(((word >> (base + 8 + h)) & 1u) ? c1[h] * inv_s : 0.0f);
        }
#pragma unroll
        for (int dt = 0; dt < 8; ++dt) {
            v16h B = *(const v16h*)(vbase + (size_t)(dt * 16 + r) * N_DIM + j0 + hi * 16);
            acc[dt] = __builtin_amdgcn_wmma_f32_16x16x32_f16(false, A, false, B,
                                                             (short)0, acc[dt], false, false);
        }
    }
    // accumulate over t into S[n][d]
#pragma unroll
    for (int dt = 0; dt < 8; ++dt) {
        float* sp = S + (size_t)(i0 + 8 * hi) * D_DIM + dt * 16 + r;
#pragma unroll
        for (int c = 0; c < 8; ++c) atomicAdd(sp + c * D_DIM, acc[dt][c]);
    }
}

// ---- kernel 6: out = S @ ffn_w^T + ffn_b via WMMA ---------------------------
__global__ __launch_bounds__(256) void k_ffn(const float* __restrict__ S,
                                             const float* __restrict__ W,
                                             const float* __restrict__ bias,
                                             float* __restrict__ out) {
    int wave = threadIdx.x >> 5, lane = threadIdx.x & 31;
    int n0 = (blockIdx.x * 8 + wave) * 16;
    int nlo = lane & 15, hi = lane >> 4, base = hi * 8;

    const float* arow = S + (size_t)(n0 + nlo) * D_DIM;
    v16h A[4];
#pragma unroll
    for (int kt = 0; kt < 4; ++kt) A[kt] = load_a_f32(arow, kt * 32, base);

#pragma unroll
    for (int et = 0; et < 8; ++et) {
        v8f acc = {};
        const float* brow = W + (size_t)(et * 16 + nlo) * D_DIM + hi * 8;
#pragma unroll
        for (int kt = 0; kt < 4; ++kt) {
            v16h B = load_b_f32(brow + kt * 32);
            acc = __builtin_amdgcn_wmma_f32_16x16x32_f16(false, A[kt], false, B,
                                                         (short)0, acc, false, false);
        }
        float bb = bias[et * 16 + nlo];
        float* op = out + (size_t)(n0 + 8 * hi) * D_DIM + et * 16 + nlo;
#pragma unroll
        for (int c = 0; c < 8; ++c) op[c * D_DIM] = acc[c] + bb;
    }
}

// ---- host side --------------------------------------------------------------
extern "C" void kernel_launch(void* const* d_in, const int* in_sizes, int n_in,
                              void* d_out, int out_size, void* d_ws, size_t ws_size,
                              hipStream_t stream) {
    const float* input_s = (const float*)d_in[0];
    const int*   adj     = (const int*)d_in[1];
    // d_in[2]=num_stock, d_in[3]=Wq, d_in[4]=bq, d_in[9]=f1_w, d_in[10]=f1_b:
    // unused (softmax over j is invariant to the per-row f1 term).
    const float* Wk  = (const float*)d_in[5];
    const float* bk  = (const float*)d_in[6];
    const float* Wv  = (const float*)d_in[7];
    const float* bv  = (const float*)d_in[8];
    const float* f2w = (const float*)d_in[11];
    const float* f2b = (const float*)d_in[12];
    const float* ffw = (const float*)d_in[13];
    const float* ffb = (const float*)d_in[14];

    char* ws = (char*)d_ws;
    float*     Sacc = (float*)ws;                                     // 1 MB
    float*     E    = (float*)(ws + (1u << 20));                      // 256 KB
    _Float16*  vT   = (_Float16*)(ws + (1u << 20) + (256u << 10));    // 16 MB
    unsigned*  mask = (unsigned*)(ws + (1u << 20) + (256u << 10) + (16u << 20)); // 16 MB

    k_zero <<<1024,   256, 0, stream>>>(Sacc);
    k_mask <<<524288, 256, 0, stream>>>(adj, mask);
    k_e    <<<256,    256, 0, stream>>>(input_s, Wk, bk, f2w, f2b, E);
    k_vproj<<<512,    256, 0, stream>>>(input_s, Wv, bv, vT);
    k_attn <<<512,    256, 0, stream>>>(E, mask, vT, Sacc);
    k_ffn  <<<16,     256, 0, stream>>>(Sacc, ffw, ffb, (float*)d_out);
}
